// VectorQuantizerEMA_88596585382087
// MI455X (gfx1250) — matrix operand order repaired and checked by reading, not compiled, not run on hardware
//
#include <hip/hip_runtime.h>
#include <hip/hip_bf16.h>

// ---------------------------------------------------------------------------
// Problem constants
// ---------------------------------------------------------------------------
#define NROWS 32768
#define DDIM  640
#define KCODE 8
#define ND    (NROWS * DDIM)          // 20,971,520 (< 2^31)

typedef __attribute__((ext_vector_type(16))) __bf16 v16bf;
typedef __attribute__((ext_vector_type(2)))  __bf16 v2bf;
typedef __attribute__((ext_vector_type(8)))  float  v8f;

union BF16x16 { v16bf v; unsigned u[8]; };

// f32 -> bf16 round-to-nearest-even
__device__ __forceinline__ unsigned short f2bf(float f) {
    unsigned u = __float_as_uint(f);
    unsigned r = u + 0x7FFFu + ((u >> 16) & 1u);
    return (unsigned short)(r >> 16);
}
__device__ __forceinline__ unsigned packbf(float a, float b) {
#if __has_builtin(__builtin_amdgcn_cvt_pk_bf16_f32)
    union { v2bf v; unsigned u; } cv;
    cv.v = __builtin_amdgcn_cvt_pk_bf16_f32(a, b);
    return cv.u;
#else
    return (unsigned)f2bf(a) | ((unsigned)f2bf(b) << 16);
#endif
}

// packed bf16 relu in one VOP3P instruction: v_pk_max_num_bf16 d, s, 0
__device__ __forceinline__ unsigned relu_pk(unsigned x) {
    unsigned r;
    asm("v_pk_max_num_bf16 %0, %1, 0" : "=v"(r) : "v"(x));
    return r;
}

__device__ __forceinline__ v8f wmma_bf16(const unsigned a[8], const unsigned b[8], v8f c) {
    BF16x16 A, B;
#pragma unroll
    for (int i = 0; i < 8; ++i) { A.u[i] = a[i]; B.u[i] = b[i]; }
    return __builtin_amdgcn_wmma_f32_16x16x32_bf16(false, A.v, false, B.v, (short)0, c, false, false);
}

// Tile geometry: rows padded to 20 dwords (80 B) -> 16-B aligned fragment
// groups and conflict-free bank mapping (gcd(20,64)=4 -> 16 distinct starts).
#define RSTRIDE 20

// A-fragment (16x32 bf16): lane<16 -> M=lane, K in {0..7,16..23}; lane>=16 -> M=lane-16, K in {8..15,24..31}
__device__ __forceinline__ void load_afrag4(const unsigned* __restrict__ row, int half, unsigned a[8]) {
    uint4 g0 = *(const uint4*)(row + half * 4);
    uint4 g1 = *(const uint4*)(row + 8 + half * 4);
    a[0] = g0.x; a[1] = g0.y; a[2] = g0.z; a[3] = g0.w;
    a[4] = g1.x; a[5] = g1.y; a[6] = g1.z; a[7] = g1.w;
}
// B-fragment (32x16 bf16): lane -> n=lane%16; lanes<16 hold K in {0..7,16..23}, lanes>=16 hold K in {8..15,24..31}
__device__ __forceinline__ void load_bfrag4(const unsigned* __restrict__ row, int khalf, unsigned b[8]) {
    uint4 g0 = *(const uint4*)(row + khalf * 4);
    uint4 g1 = *(const uint4*)(row + 8 + khalf * 4);
    b[0] = g0.x; b[1] = g0.y; b[2] = g0.z; b[3] = g0.w;
    b[4] = g1.x; b[5] = g1.y; b[6] = g1.z; b[7] = g1.w;
}

// ---------------------------------------------------------------------------
// K0: pack fc0_w / shortcut_w / fc1_w / embedding to bf16; zero accumulators
// ---------------------------------------------------------------------------
__global__ __launch_bounds__(256) void k0_pack(
    const float* __restrict__ fc0_w, const float* __restrict__ fc1_w,
    const float* __restrict__ sc_w,  const float* __restrict__ emb,
    unsigned short* __restrict__ w0b, unsigned short* __restrict__ w1b,
    unsigned short* __restrict__ wsb, unsigned short* __restrict__ embb,
    float* __restrict__ recon, int* __restrict__ cnt)
{
    long tid = (long)blockIdx.x * blockDim.x + threadIdx.x;
    if (tid == 0) *recon = 0.0f;
    if (tid < KCODE) cnt[tid] = 0;
    const long N0 = (long)DDIM * 2 * DDIM;     // 819200
    const long N1 = (long)DDIM * DDIM;         // 409600
    const long NE = (long)KCODE * DDIM;        // 5120
    const long TOT = 2 * N0 + N1 + NE;
    long stride = (long)gridDim.x * blockDim.x;
    for (long i = tid; i < TOT; i += stride) {
        if (i < N0)               w0b[i]              = f2bf(fc0_w[i]);
        else if (i < 2 * N0)      wsb[i - N0]         = f2bf(sc_w[i - N0]);
        else if (i < 2 * N0 + N1) w1b[i - 2 * N0]     = f2bf(fc1_w[i - 2 * N0]);
        else                      embb[i - 2 * N0 - N1] = f2bf(emb[i - 2 * N0 - N1]);
    }
}

// ---------------------------------------------------------------------------
// K1: VQ argmin (one wave32 per row), recon sum, counts, task_label
// ---------------------------------------------------------------------------
__global__ __launch_bounds__(256) void k1_vq(
    const float* __restrict__ cond_x, const float* __restrict__ inputs,
    const float* __restrict__ emb,
    int* __restrict__ idx_out, float* __restrict__ recon_acc,
    int* __restrict__ cnt_acc, float* __restrict__ task_label)
{
    __shared__ float semb[KCODE * DDIM];
    __shared__ float see[KCODE];
    __shared__ int   scnt[KCODE];
    int t = threadIdx.x;
    if (t < KCODE) scnt[t] = 0;
    for (int i = t; i < KCODE * DDIM; i += 256) semb[i] = emb[i];
    __syncthreads();

    int w = t >> 5, lane = t & 31;
    {
        float s = 0.f;
#pragma unroll
        for (int j = 0; j < 20; ++j) { float e = semb[w * DDIM + lane + 32 * j]; s += e * e; }
        for (int off = 16; off; off >>= 1) s += __shfl_xor(s, off, 32);
        if (lane == 0) see[w] = s;
    }
    __syncthreads();

    int row = blockIdx.x * 8 + w;
    const float* xr = inputs + row * DDIM;
    const float* cr = cond_x + row * DDIM;
    float xs[20];
    float dots[KCODE];
#pragma unroll
    for (int k = 0; k < KCODE; ++k) dots[k] = 0.f;
#pragma unroll
    for (int j = 0; j < 20; ++j) {
        int d = lane + 32 * j;
        float x = xr[d];
        xs[j] = x;
        task_label[row * DDIM + d] = x - cr[d];
#pragma unroll
        for (int k = 0; k < KCODE; ++k) dots[k] += semb[k * DDIM + d] * x;
    }
#pragma unroll
    for (int k = 0; k < KCODE; ++k)
        for (int off = 16; off; off >>= 1) dots[k] += __shfl_xor(dots[k], off, 32);

    int best = 0;
    float bs = see[0] - 2.f * dots[0];
#pragma unroll
    for (int k = 1; k < KCODE; ++k) {
        float s = see[k] - 2.f * dots[k];
        if (s < bs) { bs = s; best = k; }
    }
    float rs = 0.f;
#pragma unroll
    for (int j = 0; j < 20; ++j) {
        int d = lane + 32 * j;
        float dlt = semb[best * DDIM + d] - xs[j];
        rs += dlt * dlt;
    }
    for (int off = 16; off; off >>= 1) rs += __shfl_xor(rs, off, 32);
    if (lane == 0) {
        idx_out[row] = best;
        atomicAdd(recon_acc, rs);
        atomicAdd(&scnt[best], 1);
    }
    __syncthreads();
    if (t < KCODE && scnt[t]) atomicAdd(&cnt_acc[t], scnt[t]);
}

// ---------------------------------------------------------------------------
// K2: fused fc0 + shortcut GEMM, double-buffered LDS.
// WG macro-tile 128x128, 8 waves as 2x4 -> wave tile 64 rows x 32 cols,
// both matrices per wave: 16 WMMAs per step, 24 ds_b128 (1.5 b128/WMMA).
// Buffer layout (dwords): ARAW[128*20] ARELU[128*20] B0[128*20] B1[128*20]
// ---------------------------------------------------------------------------
#define K2_ARAW  0
#define K2_ARELU (128 * RSTRIDE)
#define K2_B0    (2 * 128 * RSTRIDE)
#define K2_B1    (3 * 128 * RSTRIDE)
#define K2_BUF   (4 * 128 * RSTRIDE)   // 10240 dwords = 40 KB per buffer

__device__ __forceinline__ void fill_k2(
    unsigned* __restrict__ buf, int t, int rowbase, int colbase, int kbase,
    const int* __restrict__ idxp, const unsigned* __restrict__ embU,
    const float* __restrict__ cond_x,
    const unsigned* __restrict__ w0U, const unsigned* __restrict__ wsU)
{
    // A tile: 128 rows x 16 dwords; thread -> (row = t/2, dword-group = (t&1)*8)
    int r = t >> 1, dg = (t & 1) * 8;
    int rowg = rowbase + r;
    unsigned v[8];
    if (kbase < DDIM) {
        int id = idxp[rowg];
        const uint4* s = (const uint4*)(embU + id * (DDIM / 2) + (kbase >> 1) + dg);
        uint4 g0 = s[0], g1 = s[1];
        v[0] = g0.x; v[1] = g0.y; v[2] = g0.z; v[3] = g0.w;
        v[4] = g1.x; v[5] = g1.y; v[6] = g1.z; v[7] = g1.w;
    } else {
        const float4* s = (const float4*)(cond_x + rowg * DDIM + (kbase - DDIM) + dg * 2);
#pragma unroll
        for (int j = 0; j < 4; ++j) {
            float4 f = s[j];
            v[2 * j]     = packbf(f.x, f.y);
            v[2 * j + 1] = packbf(f.z, f.w);
        }
    }
    unsigned* ar = buf + K2_ARAW + r * RSTRIDE + dg;
    *(uint4*)ar       = make_uint4(v[0], v[1], v[2], v[3]);
    *(uint4*)(ar + 4) = make_uint4(v[4], v[5], v[6], v[7]);
    unsigned w[8];
#pragma unroll
    for (int i = 0; i < 8; ++i) w[i] = relu_pk(v[i]);
    unsigned* al = buf + K2_ARELU + r * RSTRIDE + dg;
    *(uint4*)al       = make_uint4(w[0], w[1], w[2], w[3]);
    *(uint4*)(al + 4) = make_uint4(w[4], w[5], w[6], w[7]);

    // B tiles: 128 n-rows x 16 dwords each; thread -> (n = t/2, group = (t&1)*8)
    int ng = colbase + r;
    int src = ng * 640 + (kbase >> 1) + dg;    // weight row = 1280 bf16 = 640 dwords
    const uint4* s0 = (const uint4*)(w0U + src);
    const uint4* s1 = (const uint4*)(wsU + src);
    unsigned* b0 = buf + K2_B0 + r * RSTRIDE + dg;
    unsigned* b1 = buf + K2_B1 + r * RSTRIDE + dg;
    *(uint4*)b0       = s0[0];
    *(uint4*)(b0 + 4) = s0[1];
    *(uint4*)b1       = s1[0];
    *(uint4*)(b1 + 4) = s1[1];
}

__global__ __launch_bounds__(256) void k2_fc0_sc(
    const float* __restrict__ cond_x, const float* __restrict__ fc0_b,
    const int* __restrict__ idxp, const unsigned* __restrict__ embU,
    const unsigned* __restrict__ w0U, const unsigned* __restrict__ wsU,
    unsigned short* __restrict__ A2, float* __restrict__ te_out)
{
    __shared__ __align__(16) unsigned lds[2][K2_BUF];   // 80 KB

    const int t = threadIdx.x;
    const int lane = t & 31, w = t >> 5;
    const int Rbase = (w & 1) * 64, Cbase = (w >> 1) * 32;
    const int rowbase = blockIdx.x * 128;
    const int colbase = blockIdx.y * 128;
    const int half = lane >> 4, ln = lane & 15;

    v8f accN[4][2], accS[4][2];
#pragma unroll
    for (int i = 0; i < 4; ++i)
#pragma unroll
        for (int j = 0; j < 2; ++j) { accN[i][j] = (v8f)0.0f; accS[i][j] = (v8f)0.0f; }

    fill_k2(lds[0], t, rowbase, colbase, 0, idxp, embU, cond_x, w0U, wsU);
    __syncthreads();

    int cur = 0;
    for (int ks = 0; ks < 40; ++ks) {
        if (ks + 1 < 40)
            fill_k2(lds[cur ^ 1], t, rowbase, colbase, (ks + 1) * 32,
                    idxp, embU, cond_x, w0U, wsU);

        const unsigned* buf = lds[cur];
        unsigned b00[8], b01[8], b10[8], b11[8];
        load_bfrag4(buf + K2_B0 + (Cbase + ln) * RSTRIDE,      half, b00);
        load_bfrag4(buf + K2_B0 + (Cbase + 16 + ln) * RSTRIDE, half, b01);
        load_bfrag4(buf + K2_B1 + (Cbase + ln) * RSTRIDE,      half, b10);
        load_bfrag4(buf + K2_B1 + (Cbase + 16 + ln) * RSTRIDE, half, b11);
#pragma unroll
        for (int mt = 0; mt < 4; ++mt) {
            unsigned a[8], rr[8];
            load_afrag4(buf + K2_ARAW  + (Rbase + mt * 16 + ln) * RSTRIDE, half, a);
            load_afrag4(buf + K2_ARELU + (Rbase + mt * 16 + ln) * RSTRIDE, half, rr);
            accN[mt][0] = wmma_bf16(rr, b00, accN[mt][0]);
            accN[mt][1] = wmma_bf16(rr, b01, accN[mt][1]);
            accS[mt][0] = wmma_bf16(a,  b10, accS[mt][0]);
            accS[mt][1] = wmma_bf16(a,  b11, accS[mt][1]);
        }
        __syncthreads();
        cur ^= 1;
    }

    // epilogue: net=relu(acc+b) -> bf16 scratch; shortcut -> f32 te region
#pragma unroll
    for (int mt = 0; mt < 4; ++mt)
#pragma unroll
        for (int nt = 0; nt < 2; ++nt) {
            int colg = colbase + Cbase + nt * 16 + ln;
            float bias = fc0_b[colg];
#pragma unroll
            for (int r = 0; r < 8; ++r) {
                int rowg = rowbase + Rbase + mt * 16 + r + 8 * half;
                float net = accN[mt][nt][r] + bias;
                net = net > 0.f ? net : 0.f;
                A2[rowg * DDIM + colg] = f2bf(net);
                te_out[rowg * DDIM + colg] = accS[mt][nt][r];
            }
        }
}

// ---------------------------------------------------------------------------
// K3: fc1 GEMM, double-buffered.  WG 128x128, wave tile 64x32.
// Buffer: A[128*20] B[128*20]
// ---------------------------------------------------------------------------
#define K3_A   0
#define K3_B   (128 * RSTRIDE)
#define K3_BUF (2 * 128 * RSTRIDE)   // 5120 dwords = 20 KB per buffer

__device__ __forceinline__ void fill_k3(
    unsigned* __restrict__ buf, int t, int rowbase, int colbase, int kbase,
    const unsigned* __restrict__ A2U, const unsigned* __restrict__ w1U)
{
    int r = t >> 1, dg = (t & 1) * 8;
    int rowg = rowbase + r;
    const uint4* s = (const uint4*)(A2U + rowg * (DDIM / 2) + (kbase >> 1) + dg);
    uint4 g0 = s[0], g1 = s[1];
    unsigned* a = buf + K3_A + r * RSTRIDE + dg;
    *(uint4*)a       = g0;
    *(uint4*)(a + 4) = g1;

    int ng = colbase + r;
    const uint4* sb = (const uint4*)(w1U + ng * (DDIM / 2) + (kbase >> 1) + dg);
    unsigned* b = buf + K3_B + r * RSTRIDE + dg;
    *(uint4*)b       = sb[0];
    *(uint4*)(b + 4) = sb[1];
}

__global__ __launch_bounds__(256) void k3_fc1(
    const unsigned* __restrict__ A2U, const unsigned* __restrict__ w1U,
    const float* __restrict__ fc1_b, float* __restrict__ te_out)
{
    __shared__ __align__(16) unsigned lds[2][K3_BUF];   // 40 KB

    const int t = threadIdx.x;
    const int lane = t & 31, w = t >> 5;
    const int Rbase = (w & 1) * 64, Cbase = (w >> 1) * 32;
    const int rowbase = blockIdx.x * 128;
    const int colbase = blockIdx.y * 128;
    const int half = lane >> 4, ln = lane & 15;

    v8f acc[4][2];
#pragma unroll
    for (int i = 0; i < 4; ++i)
#pragma unroll
        for (int j = 0; j < 2; ++j) acc[i][j] = (v8f)0.0f;

    fill_k3(lds[0], t, rowbase, colbase, 0, A2U, w1U);
    __syncthreads();

    int cur = 0;
    for (int ks = 0; ks < 20; ++ks) {
        if (ks + 1 < 20)
            fill_k3(lds[cur ^ 1], t, rowbase, colbase, (ks + 1) * 32, A2U, w1U);

        const unsigned* buf = lds[cur];
        unsigned b0[8], b1[8];
        load_bfrag4(buf + K3_B + (Cbase + ln) * RSTRIDE,      half, b0);
        load_bfrag4(buf + K3_B + (Cbase + 16 + ln) * RSTRIDE, half, b1);
#pragma unroll
        for (int mt = 0; mt < 4; ++mt) {
            unsigned a[8];
            load_afrag4(buf + K3_A + (Rbase + mt * 16 + ln) * RSTRIDE, half, a);
            acc[mt][0] = wmma_bf16(a, b0, acc[mt][0]);
            acc[mt][1] = wmma_bf16(a, b1, acc[mt][1]);
        }
        __syncthreads();
        cur ^= 1;
    }

#pragma unroll
    for (int mt = 0; mt < 4; ++mt)
#pragma unroll
        for (int nt = 0; nt < 2; ++nt) {
            int colg = colbase + Cbase + nt * 16 + ln;
            float bias = fc1_b[colg];
#pragma unroll
            for (int r = 0; r < 8; ++r) {
                int rowg = rowbase + Rbase + mt * 16 + r + 8 * half;
                int o = rowg * DDIM + colg;
                te_out[o] = te_out[o] + acc[mt][nt][r] + bias;
            }
        }
}

// ---------------------------------------------------------------------------
// K4: elementwise loss + perplexity scalar
// out layout: [loss ND][task_embed ND][perplexity 1][task_label ND]
// ---------------------------------------------------------------------------
__global__ __launch_bounds__(256) void k4_epilogue(
    const float* __restrict__ recon_acc, const int* __restrict__ cnt,
    float* __restrict__ out)
{
    int i = blockIdx.x * 256 + threadIdx.x;
    float recon = 1.25f * (*recon_acc) / (float)ND;
    float te = out[ND + i];
    float tl = out[2 * ND + 1 + i];
    float d = te - tl;
    out[i] = 10.0f * d * d + recon;
    if (i == 0) {
        float s = 0.f;
#pragma unroll
        for (int k = 0; k < KCODE; ++k) {
            float p = (float)cnt[k] / (float)NROWS;
            s += p * __logf(p + 1e-10f);
        }
        out[2 * ND] = __expf(-s);
    }
}

// ---------------------------------------------------------------------------
// Launcher
// ---------------------------------------------------------------------------
extern "C" void kernel_launch(void* const* d_in, const int* in_sizes, int n_in,
                              void* d_out, int out_size, void* d_ws, size_t ws_size,
                              hipStream_t stream) {
    const float* cond_x     = (const float*)d_in[0];
    const float* inputs     = (const float*)d_in[1];
    const float* embedding  = (const float*)d_in[2];
    const float* fc0_w      = (const float*)d_in[3];
    const float* fc0_b      = (const float*)d_in[4];
    const float* fc1_w      = (const float*)d_in[5];
    const float* fc1_b      = (const float*)d_in[6];
    const float* shortcut_w = (const float*)d_in[7];
    float* out = (float*)d_out;

    // workspace layout (bytes)
    char* ws = (char*)d_ws;
    float*          recon = (float*)ws;                      // 4 B
    int*            cnt   = (int*)(ws + 4);                  // 32 B
    int*            idx   = (int*)(ws + 64);                 // 128 KB
    unsigned short* embb  = (unsigned short*)(ws + 131200);  // 10 KB
    unsigned short* w0b   = (unsigned short*)(ws + 141440);  // 1.56 MB
    unsigned short* wsb   = (unsigned short*)(ws + 1779840); // 1.56 MB
    unsigned short* w1b   = (unsigned short*)(ws + 3418240); // 0.78 MB
    unsigned short* A2    = (unsigned short*)(ws + 4237440); // 40 MB

    k0_pack<<<2048, 256, 0, stream>>>(fc0_w, fc1_w, shortcut_w, embedding,
                                      w0b, w1b, wsb, embb, recon, cnt);
    k1_vq<<<NROWS / 8, 256, 0, stream>>>(cond_x, inputs, embedding,
                                         idx, recon, cnt, out + 2L * ND + 1);
    k2_fc0_sc<<<dim3(NROWS / 128, DDIM / 128), 256, 0, stream>>>(
        cond_x, fc0_b, idx, (const unsigned*)embb,
        (const unsigned*)w0b, (const unsigned*)wsb, A2, out + ND);
    k3_fc1<<<dim3(NROWS / 128, DDIM / 128), 256, 0, stream>>>(
        (const unsigned*)A2, (const unsigned*)w1b, fc1_b, out + ND);
    k4_epilogue<<<ND / 256, 256, 0, stream>>>(recon, cnt, out);
}
